// GeometricESAModel_36301063586215
// MI455X (gfx1250) — matrix operand order repaired and checked by reading, not compiled, not run on hardware
//
#include <hip/hip_runtime.h>
#include <hip/hip_bf16.h>

#define HD    128
#define NATOM 16384
#define NEDGE 262144
#define NBLKS 1024
#define BSZ   16
#define LBLK  64
#define EBE   512
#define KEF   288   // 262 padded to multiple of 32

typedef __attribute__((ext_vector_type(16))) _Float16 v16h;
typedef __attribute__((ext_vector_type(8)))  _Float16 v8h;
typedef __attribute__((ext_vector_type(8)))  float    v8f;

// ---------------- CDNA5 async global->LDS copy (ASYNCcnt path) ----------------
#if defined(__gfx1250__) && __has_builtin(__builtin_amdgcn_global_load_async_to_lds_b128)
#define HAVE_ASYNC_LDS 1
#else
#define HAVE_ASYNC_LDS 0
#endif

#if HAVE_ASYNC_LDS
typedef int b128i __attribute__((vector_size(16)));          // V4i (GCC vector, matches builtin)
typedef __attribute__((address_space(1))) b128i b128_glb;    // global
typedef __attribute__((address_space(3))) b128i b128_lds;    // LDS
#endif

// 16-byte global->LDS copy; async (no VGPR data path) when available.
__device__ __forceinline__ void cp_async16(const _Float16* g, _Float16* l) {
#if HAVE_ASYNC_LDS
  __builtin_amdgcn_global_load_async_to_lds_b128((b128_glb*)g, (b128_lds*)l, 0, 0);
#else
  *(v8h*)l = *(const v8h*)g;
#endif
}

__device__ __forceinline__ void async_fence_barrier() {
#if HAVE_ASYNC_LDS
#if __has_builtin(__builtin_amdgcn_s_wait_asynccnt)
  __builtin_amdgcn_s_wait_asynccnt(0);
#else
  asm volatile("s_wait_asynccnt 0x0" ::: "memory");
#endif
#endif
  __syncthreads();
}

// ---------------- WMMA helpers ----------------

__device__ __forceinline__ v8f zero8() {
  v8f z;
#pragma unroll
  for (int i = 0; i < 8; ++i) z[i] = 0.f;
  return z;
}

__device__ __forceinline__ v16h make_frag(v8h lo, v8h hi) {
  v16h f;
#pragma unroll
  for (int t = 0; t < 8; ++t) { f[t] = lo[t]; f[t + 8] = hi[t]; }
  return f;
}

// A fragment (16x32 f16): rows m0..m0+15 of LDS tile with row stride ld (halves),
// K chunk starting at k0. Lane L: m = m0 + L%16, holds K = k0+(L/16)*8 .. +7 and +16..+23.
__device__ __forceinline__ v16h a_frag_lds(const _Float16* a, int ld, int m0, int k0, int lane) {
  int m  = m0 + (lane & 15);
  int kb = k0 + ((lane >> 4) << 3);
  const _Float16* p = a + (size_t)m * ld + kb;
  return make_frag(*(const v8h*)p, *(const v8h*)(p + 16));
}

// B fragment (32x16 f16) from global Bt stored [N][Kld] (K contiguous).
// Lane L: n = n0 + L%16, holds K = k0+(L/16)*8 .. +7 and +16..+23.
__device__ __forceinline__ v16h b_frag_glb(const _Float16* bt, int ld, int n0, int k0, int lane) {
  int n = n0 + (lane & 15);
  int k = k0 + ((lane >> 4) << 3);
  const _Float16* p = bt + (size_t)n * ld + k;
  return make_frag(*(const v8h*)p, *(const v8h*)(p + 16));
}

__device__ __forceinline__ v8f wmma16(v16h a, v16h b, v8f c) {
  return __builtin_amdgcn_wmma_f32_16x16x32_f16(false, a, false, b, (short)0, c, false, false);
}

__device__ __forceinline__ float silu(float v) { return v / (1.f + __expf(-v)); }

// ---------------- small utility kernels ----------------

__global__ void k_zero(float* __restrict__ p, int n) {
  int t = blockIdx.x * 256 + threadIdx.x;
  if (t < n) p[t] = 0.f;
}

// transpose+convert fp32 [K][N] -> f16 [N][Kpad] (zero padded)
__global__ void k_convert_t(const float* __restrict__ src, _Float16* __restrict__ dst,
                            int K, int N, int Kpad) {
  int t = blockIdx.x * 256 + threadIdx.x;
  if (t >= N * Kpad) return;
  int n = t / Kpad, k = t - n * Kpad;
  dst[t] = (k < K) ? (_Float16)src[(size_t)k * N + n] : (_Float16)0.f;
}

__global__ void k_embed(const int* __restrict__ A, const int* __restrict__ pos,
                        const int* __restrict__ btypes,
                        const float* __restrict__ emb_atom, const float* __restrict__ emb_pos,
                        const float* __restrict__ emb_block,
                        float* __restrict__ x, _Float16* __restrict__ xh) {
  int t = blockIdx.x * 256 + threadIdx.x;
  if (t >= NATOM * HD) return;
  int n = t >> 7, h = t & 127;
  float v = emb_atom[A[n] * HD + h] + emb_pos[pos[n] * HD + h] +
            emb_block[btypes[n >> 4] * HD + h];
  x[t] = v;
  xh[t] = (_Float16)v;
}

__global__ void k_rbf(const float* __restrict__ coords, const int* __restrict__ ei,
                      const int* __restrict__ ej, const float* __restrict__ freq,
                      _Float16* __restrict__ rbfh) {
  int e = blockIdx.x * 256 + threadIdx.x;
  if (e >= NEDGE) return;
  int i = ei[e], j = ej[e];
  float dx = coords[i * 3 + 0] - coords[j * 3 + 0];
  float dy = coords[i * 3 + 1] - coords[j * 3 + 1];
  float dz = coords[i * 3 + 2] - coords[j * 3 + 2];
  float d = sqrtf(dx * dx + dy * dy + dz * dz);
  float u = fmaxf(d * 0.125f, 1e-3f);
  float u2 = u * u, u4 = u2 * u2, u5 = u4 * u;
  float env = 1.f / u - 28.f * u5 + 48.f * u5 * u - 21.f * u5 * u2;
#pragma unroll
  for (int r = 0; r < 6; ++r)
    rbfh[(size_t)e * 8 + r] = (_Float16)(env * __sinf(freq[r] * u));
  rbfh[(size_t)e * 8 + 6] = (_Float16)0.f;
  rbfh[(size_t)e * 8 + 7] = (_Float16)0.f;
}

__global__ void k_pool(const float* __restrict__ x, float* __restrict__ blocks,
                       _Float16* __restrict__ blocksh) {
  int t = blockIdx.x * 256 + threadIdx.x;
  if (t >= NBLKS * HD) return;
  int nb = t >> 7, h = t & 127;
  float s = 0.f;
#pragma unroll
  for (int k = 0; k < 16; ++k) s += x[(size_t)((nb << 4) + k) * HD + h];
  s *= (1.f / 16.f);
  blocks[t] = s;
  blocksh[t] = (_Float16)s;
}

// ---------------- edge message MLP (the 52-GFLOP hot loop) ----------------
// out: agg[i[e]] += silu([x_i|x_j|rbf] @ W1 + b1) @ W2 + b2
__global__ __launch_bounds__(256) void k_edge_mlp(
    const _Float16* __restrict__ xh, const _Float16* __restrict__ rbfh,
    const int* __restrict__ ei, const int* __restrict__ ej,
    const _Float16* __restrict__ w1t, const float* __restrict__ b1,
    const _Float16* __restrict__ w2t, const float* __restrict__ b2,
    float* __restrict__ agg) {
  __shared__ __align__(16) _Float16 ef[64 * KEF];
  __shared__ __align__(16) _Float16 hb[64 * HD];
  __shared__ int si[64], sj[64];
  const int tid = threadIdx.x;
  const int e0 = blockIdx.x * 64;
  if (tid < 64) { si[tid] = ei[e0 + tid]; sj[tid] = ej[e0 + tid]; }
  __syncthreads();
  // rbf + zero-pad tail columns (scalar; tiny)
  for (int t = tid; t < 64 * 32; t += 256) {
    int r = t >> 5, c = 256 + (t & 31);
    ef[r * KEF + c] = (c < 264) ? rbfh[(size_t)(e0 + r) * 8 + (c - 256)]
                                : (_Float16)0.f;
  }
  // x_i | x_j gather: per-row 256B contiguous -> async 16B global->LDS copies
  for (int t = tid; t < 64 * 16; t += 256) {
    int r = t >> 4, c8 = (t & 15) << 3;
    cp_async16(xh + (size_t)si[r] * HD + c8, ef + r * KEF + c8);
    cp_async16(xh + (size_t)sj[r] * HD + c8, ef + r * KEF + HD + c8);
  }
  async_fence_barrier();
  const int wave = tid >> 5, lane = tid & 31;
  const int n0 = wave * 16;
  const int nc = n0 + (lane & 15);
  const int rr = (lane >> 4) << 3;
  // GEMM1: [64 x 288] @ [288 x 128] -> hidden, silu
  {
    float bb = b1[nc];
    v8f acc[4];
#pragma unroll
    for (int m = 0; m < 4; ++m) acc[m] = zero8();
    for (int kc = 0; kc < KEF; kc += 32) {
      v16h bf = b_frag_glb(w1t, KEF, n0, kc, lane);
      v16h af[4];
#pragma unroll
      for (int mt = 0; mt < 4; ++mt) af[mt] = a_frag_lds(ef, KEF, mt * 16, kc, lane);
#pragma unroll
      for (int mt = 0; mt < 4; ++mt) acc[mt] = wmma16(af[mt], bf, acc[mt]);
    }
#pragma unroll
    for (int mt = 0; mt < 4; ++mt)
#pragma unroll
      for (int r = 0; r < 8; ++r) {
        int m = mt * 16 + r + rr;
        hb[(size_t)m * HD + nc] = (_Float16)silu(acc[mt][r] + bb);
      }
  }
  __syncthreads();
  // GEMM2: [64 x 128] @ [128 x 128] -> message, scatter-add
  {
    float bb = b2[nc];
    v8f acc[4];
#pragma unroll
    for (int m = 0; m < 4; ++m) acc[m] = zero8();
    for (int kc = 0; kc < HD; kc += 32) {
      v16h bf = b_frag_glb(w2t, HD, n0, kc, lane);
      v16h af[4];
#pragma unroll
      for (int mt = 0; mt < 4; ++mt) af[mt] = a_frag_lds(hb, HD, mt * 16, kc, lane);
#pragma unroll
      for (int mt = 0; mt < 4; ++mt) acc[mt] = wmma16(af[mt], bf, acc[mt]);
    }
#pragma unroll
    for (int mt = 0; mt < 4; ++mt)
#pragma unroll
      for (int r = 0; r < 8; ++r) {
        int m = mt * 16 + r + rr;
        atomicAdd(&agg[(size_t)si[m] * HD + nc], acc[mt][r] + bb);
      }
  }
}

// ---------------- atom update MLP: x = silu([x|agg]@W1+b1)@W2+b2 + x ----------------
__global__ __launch_bounds__(256) void k_update(
    const float* __restrict__ aggv,
    const _Float16* __restrict__ w1t, const float* __restrict__ b1,
    const _Float16* __restrict__ w2t, const float* __restrict__ b2,
    float* __restrict__ x, _Float16* __restrict__ xh) {
  __shared__ __align__(16) _Float16 ui[64 * 256];
  __shared__ __align__(16) _Float16 hb[64 * HD];
  const int tid = threadIdx.x;
  const int a0 = blockIdx.x * 64;
  // x half: contiguous f16 -> async copies
  for (int t = tid; t < 64 * 16; t += 256) {
    int r = t >> 4, c8 = (t & 15) << 3;
    cp_async16(xh + (size_t)(a0 + r) * HD + c8, ui + r * 256 + c8);
  }
  // agg half: f32 -> f16 conversion (must go through VALU)
  for (int t = tid; t < 64 * HD; t += 256) {
    int r = t >> 7, c = t & 127;
    ui[r * 256 + HD + c] = (_Float16)aggv[(size_t)(a0 + r) * HD + c];
  }
  async_fence_barrier();
  const int wave = tid >> 5, lane = tid & 31;
  const int n0 = wave * 16;
  const int nc = n0 + (lane & 15);
  const int rr = (lane >> 4) << 3;
  {
    float bb = b1[nc];
    v8f acc[4];
#pragma unroll
    for (int m = 0; m < 4; ++m) acc[m] = zero8();
    for (int kc = 0; kc < 256; kc += 32) {
      v16h bf = b_frag_glb(w1t, 256, n0, kc, lane);
      v16h af[4];
#pragma unroll
      for (int mt = 0; mt < 4; ++mt) af[mt] = a_frag_lds(ui, 256, mt * 16, kc, lane);
#pragma unroll
      for (int mt = 0; mt < 4; ++mt) acc[mt] = wmma16(af[mt], bf, acc[mt]);
    }
#pragma unroll
    for (int mt = 0; mt < 4; ++mt)
#pragma unroll
      for (int r = 0; r < 8; ++r) {
        int m = mt * 16 + r + rr;
        hb[(size_t)m * HD + nc] = (_Float16)silu(acc[mt][r] + bb);
      }
  }
  __syncthreads();
  {
    float bb = b2[nc];
    v8f acc[4];
#pragma unroll
    for (int m = 0; m < 4; ++m) acc[m] = zero8();
    for (int kc = 0; kc < HD; kc += 32) {
      v16h bf = b_frag_glb(w2t, HD, n0, kc, lane);
      v16h af[4];
#pragma unroll
      for (int mt = 0; mt < 4; ++mt) af[mt] = a_frag_lds(hb, HD, mt * 16, kc, lane);
#pragma unroll
      for (int mt = 0; mt < 4; ++mt) acc[mt] = wmma16(af[mt], bf, acc[mt]);
    }
#pragma unroll
    for (int mt = 0; mt < 4; ++mt)
#pragma unroll
      for (int r = 0; r < 8; ++r) {
        int m = mt * 16 + r + rr;
        size_t idx = (size_t)(a0 + m) * HD + nc;
        float v = acc[mt][r] + bb + x[idx];   // residual
        x[idx] = v;
        xh[idx] = (_Float16)v;
      }
  }
}

// ---------------- block-edge feature MLP: relu([b_src|b_dst]@W1+b1)@W2+b2 ----------------
__global__ __launch_bounds__(256) void k_blockedge_mlp(
    const _Float16* __restrict__ blocksh, const int* __restrict__ be,
    const _Float16* __restrict__ w1t, const float* __restrict__ b1,
    const _Float16* __restrict__ w2t, const float* __restrict__ b2,
    _Float16* __restrict__ ef2h) {
  __shared__ __align__(16) _Float16 ui[64 * 256];
  __shared__ __align__(16) _Float16 hb[64 * HD];
  __shared__ int ss[64], sd[64];
  const int tid = threadIdx.x;
  const int e0 = blockIdx.x * 64;
  if (tid < 64) {
    int row = e0 + tid;
    int g = row >> 9, eb = row & 511;
    ss[tid] = g * LBLK + be[(g * EBE + eb) * 2 + 0];
    sd[tid] = g * LBLK + be[(g * EBE + eb) * 2 + 1];
  }
  __syncthreads();
  for (int t = tid; t < 64 * 32; t += 256) {
    int r = t >> 5, q = t & 31;
    int part = q >> 4, c8 = (q & 15) << 3;
    int srow = part ? sd[r] : ss[r];
    cp_async16(blocksh + (size_t)srow * HD + c8, ui + r * 256 + part * HD + c8);
  }
  async_fence_barrier();
  const int wave = tid >> 5, lane = tid & 31;
  const int n0 = wave * 16;
  const int nc = n0 + (lane & 15);
  const int rr = (lane >> 4) << 3;
  {
    float bb = b1[nc];
    v8f acc[4];
#pragma unroll
    for (int m = 0; m < 4; ++m) acc[m] = zero8();
    for (int kc = 0; kc < 256; kc += 32) {
      v16h bf = b_frag_glb(w1t, 256, n0, kc, lane);
      v16h af[4];
#pragma unroll
      for (int mt = 0; mt < 4; ++mt) af[mt] = a_frag_lds(ui, 256, mt * 16, kc, lane);
#pragma unroll
      for (int mt = 0; mt < 4; ++mt) acc[mt] = wmma16(af[mt], bf, acc[mt]);
    }
#pragma unroll
    for (int mt = 0; mt < 4; ++mt)
#pragma unroll
      for (int r = 0; r < 8; ++r) {
        int m = mt * 16 + r + rr;
        hb[(size_t)m * HD + nc] = (_Float16)fmaxf(acc[mt][r] + bb, 0.f);
      }
  }
  __syncthreads();
  {
    float bb = b2[nc];
    v8f acc[4];
#pragma unroll
    for (int m = 0; m < 4; ++m) acc[m] = zero8();
    for (int kc = 0; kc < HD; kc += 32) {
      v16h bf = b_frag_glb(w2t, HD, n0, kc, lane);
      v16h af[4];
#pragma unroll
      for (int mt = 0; mt < 4; ++mt) af[mt] = a_frag_lds(hb, HD, mt * 16, kc, lane);
#pragma unroll
      for (int mt = 0; mt < 4; ++mt) acc[mt] = wmma16(af[mt], bf, acc[mt]);
    }
#pragma unroll
    for (int mt = 0; mt < 4; ++mt)
#pragma unroll
      for (int r = 0; r < 8; ++r) {
        int m = mt * 16 + r + rr;
        ef2h[(size_t)(e0 + m) * HD + nc] = (_Float16)(acc[mt][r] + bb);
      }
  }
}

// ---------------- generic dense GEMM: C[M][N] = A[M][K=128] @ Bt^T + bias ----------------
__global__ __launch_bounds__(256) void k_gemm(
    const _Float16* __restrict__ A, const _Float16* __restrict__ Bt,
    const float* __restrict__ bias, _Float16* __restrict__ Ch,
    float* __restrict__ Cf, int N, int K) {
  __shared__ __align__(16) _Float16 at[64 * 128];
  const int tid = threadIdx.x;
  const int r0 = blockIdx.x * 64;
  for (int t = tid; t < (64 * K) >> 3; t += 256)
    cp_async16(A + (size_t)r0 * K + ((size_t)t << 3), at + (t << 3));
  async_fence_barrier();
  const int wave = tid >> 5, lane = tid & 31;
  const int rr = (lane >> 4) << 3;
  for (int nt = wave; nt * 16 < N; nt += 8) {
    int n0 = nt * 16;
    int nc = n0 + (lane & 15);
    float bb = bias[nc];
    v8f acc[4];
#pragma unroll
    for (int m = 0; m < 4; ++m) acc[m] = zero8();
    for (int kc = 0; kc < K; kc += 32) {
      v16h bf = b_frag_glb(Bt, K, n0, kc, lane);
      v16h af[4];
#pragma unroll
      for (int mt = 0; mt < 4; ++mt) af[mt] = a_frag_lds(at, K, mt * 16, kc, lane);
#pragma unroll
      for (int mt = 0; mt < 4; ++mt) acc[mt] = wmma16(af[mt], bf, acc[mt]);
    }
#pragma unroll
    for (int mt = 0; mt < 4; ++mt)
#pragma unroll
      for (int r = 0; r < 8; ++r) {
        int m = mt * 16 + r + rr;
        size_t idx = (size_t)(r0 + m) * N + nc;
        float v = acc[mt][r] + bb;
        if (Ch) Ch[idx] = (_Float16)v;
        if (Cf) Cf[idx] = v;
      }
  }
}

// ---------------- multi-head attention core (dh=16, NH=8) ----------------
// one workgroup per (graph, head); K/V tiles in LDS; max-subtracted softmax
__global__ __launch_bounds__(256) void k_attn(const _Float16* __restrict__ qkv,
                                              _Float16* __restrict__ oh, int S) {
  __shared__ __align__(16) _Float16 ks[512 * 16];
  __shared__ __align__(16) _Float16 vs[512 * 16];
  const int g = blockIdx.x >> 3, head = blockIdx.x & 7;
  const int rb = g * S;
  const int tid = threadIdx.x;
  for (int t = tid; t < S * 2; t += 256) {
    int s = t >> 1, hh = (t & 1) << 3;
    cp_async16(qkv + (size_t)(rb + s) * 384 + 128 + head * 16 + hh, ks + s * 16 + hh);
    cp_async16(qkv + (size_t)(rb + s) * 384 + 256 + head * 16 + hh, vs + s * 16 + hh);
  }
  async_fence_barrier();
  for (int q = tid; q < S; q += 256) {
    float qv[16];
#pragma unroll
    for (int c = 0; c < 16; ++c)
      qv[c] = 0.25f * (float)qkv[(size_t)(rb + q) * 384 + head * 16 + c]; // 1/sqrt(16)
    float mx = -1e30f;
    for (int s = 0; s < S; ++s) {
      float d = 0.f;
#pragma unroll
      for (int c = 0; c < 16; ++c) d += qv[c] * (float)ks[s * 16 + c];
      mx = fmaxf(mx, d);
    }
    float den = 0.f;
    float o[16];
#pragma unroll
    for (int c = 0; c < 16; ++c) o[c] = 0.f;
    for (int s = 0; s < S; ++s) {
      float d = 0.f;
#pragma unroll
      for (int c = 0; c < 16; ++c) d += qv[c] * (float)ks[s * 16 + c];
      float w = __expf(d - mx);
      den += w;
#pragma unroll
      for (int c = 0; c < 16; ++c) o[c] += w * (float)vs[s * 16 + c];
    }
    float inv = 1.f / den;
#pragma unroll
    for (int c = 0; c < 16; ++c)
      oh[(size_t)(rb + q) * HD + head * 16 + c] = (_Float16)(o[c] * inv);
  }
}

// ---------------- block-edge aggregation (segment mean) ----------------
__global__ void k_cnt(const int* __restrict__ be, float* __restrict__ cnt) {
  int t = blockIdx.x * 256 + threadIdx.x;
  if (t >= BSZ * EBE) return;
  int g = t >> 9, eb = t & 511;
  int src = be[(g * EBE + eb) * 2 + 0];
  atomicAdd(&cnt[g * LBLK + src], 1.f);
}

__global__ void k_sums(const int* __restrict__ be, const float* __restrict__ ea,
                       float* __restrict__ sums) {
  int t = blockIdx.x * 256 + threadIdx.x;
  if (t >= BSZ * EBE * HD) return;
  int row = t >> 7, h = t & 127;
  int g = row >> 9, eb = row & 511;
  int src = be[(g * EBE + eb) * 2 + 0];
  atomicAdd(&sums[(size_t)(g * LBLK + src) * HD + h], ea[t]);
}

__global__ void k_blockadd(const float* __restrict__ sums, const float* __restrict__ cnt,
                           float* __restrict__ blocks, _Float16* __restrict__ blocksh) {
  int t = blockIdx.x * 256 + threadIdx.x;
  if (t >= NBLKS * HD) return;
  float v = blocks[t] + sums[t] / fmaxf(cnt[t >> 7], 1.f);
  blocks[t] = v;
  blocksh[t] = (_Float16)v;
}

// ---------------- mean-pool + layernorm + final MLP ----------------
__global__ __launch_bounds__(128) void k_final(
    const float* __restrict__ att, const float* __restrict__ lng, const float* __restrict__ lnb,
    const float* __restrict__ w1, const float* __restrict__ b1,
    const float* __restrict__ w2, const float* __restrict__ b2,
    const float* __restrict__ w3, const float* __restrict__ b3, float* __restrict__ out) {
  __shared__ float gv[128], t1[128], red[128];
  const int g = blockIdx.x, h = threadIdx.x;
  float s = 0.f;
  for (int l = 0; l < LBLK; ++l) s += att[(size_t)(g * LBLK + l) * HD + h];
  s *= (1.f / (float)LBLK);
  red[h] = s; __syncthreads();
  for (int o = 64; o > 0; o >>= 1) { if (h < o) red[h] += red[h + o]; __syncthreads(); }
  float mu = red[0] * (1.f / 128.f); __syncthreads();
  float dd = s - mu;
  red[h] = dd * dd; __syncthreads();
  for (int o = 64; o > 0; o >>= 1) { if (h < o) red[h] += red[h + o]; __syncthreads(); }
  float var = red[0] * (1.f / 128.f);
  gv[h] = dd * rsqrtf(var + 1e-5f) * lng[h] + lnb[h];
  __syncthreads();
  float a1 = b1[h];
  for (int k = 0; k < 128; ++k) a1 += gv[k] * w1[k * HD + h];
  a1 = fmaxf(a1, 0.f);
  t1[h] = a1; __syncthreads();
  float a2 = b2[h];
  for (int k = 0; k < 128; ++k) a2 += t1[k] * w2[k * HD + h];
  a2 = fmaxf(a2, 0.f);
  __syncthreads();
  red[h] = a2 * w3[h]; __syncthreads();
  for (int o = 64; o > 0; o >>= 1) { if (h < o) red[h] += red[h + o]; __syncthreads(); }
  if (h == 0) out[g] = red[0] + b3[0];
}

// ---------------- workspace layout ----------------
static constexpr size_t au(size_t x) { return (x + 255) & ~(size_t)255; }
static constexpr size_t OFF_X       = 0;
static constexpr size_t OFF_XH      = OFF_X       + au((size_t)NATOM * HD * 4);
static constexpr size_t OFF_AGG     = OFF_XH      + au((size_t)NATOM * HD * 2);
static constexpr size_t OFF_RBFH    = OFF_AGG     + au((size_t)NATOM * HD * 4);
static constexpr size_t OFF_W1T     = OFF_RBFH    + au((size_t)NEDGE * 8 * 2);
static constexpr size_t OFF_W2T     = OFF_W1T     + au((size_t)2 * 128 * KEF * 2);
static constexpr size_t OFF_UW1T    = OFF_W2T     + au((size_t)2 * 128 * 128 * 2);
static constexpr size_t OFF_UW2T    = OFF_UW1T    + au((size_t)2 * 128 * 256 * 2);
static constexpr size_t OFF_EW1T    = OFF_UW2T    + au((size_t)2 * 128 * 128 * 2);
static constexpr size_t OFF_EW2T    = OFF_EW1T    + au((size_t)128 * 256 * 2);
static constexpr size_t OFF_AIWT    = OFF_EW2T    + au((size_t)128 * 128 * 2);
static constexpr size_t OFF_AOWT    = OFF_AIWT    + au((size_t)384 * 128 * 2);
static constexpr size_t OFF_BLOCKS  = OFF_AOWT    + au((size_t)128 * 128 * 2);
static constexpr size_t OFF_BLOCKSH = OFF_BLOCKS  + au((size_t)NBLKS * HD * 4);
static constexpr size_t OFF_EF2H    = OFF_BLOCKSH + au((size_t)NBLKS * HD * 2);
static constexpr size_t OFF_QKVH    = OFF_EF2H    + au((size_t)BSZ * EBE * HD * 2);
static constexpr size_t OFF_OH      = OFF_QKVH    + au((size_t)BSZ * EBE * 384 * 2);
static constexpr size_t OFF_EA      = OFF_OH      + au((size_t)BSZ * EBE * HD * 2);
static constexpr size_t OFF_SUMS    = OFF_EA      + au((size_t)BSZ * EBE * HD * 4);
static constexpr size_t OFF_CNT     = OFF_SUMS    + au((size_t)NBLKS * HD * 4);
static constexpr size_t OFF_QKV2H   = OFF_CNT     + au((size_t)NBLKS * 4);
static constexpr size_t OFF_OH2     = OFF_QKV2H   + au((size_t)NBLKS * 384 * 2);
static constexpr size_t OFF_ATT     = OFF_OH2     + au((size_t)NBLKS * HD * 2);

extern "C" void kernel_launch(void* const* d_in, const int* in_sizes, int n_in,
                              void* d_out, int out_size, void* d_ws, size_t ws_size,
                              hipStream_t stream) {
  (void)in_sizes; (void)n_in; (void)out_size; (void)ws_size;
  const int*   A         = (const int*)d_in[0];
  const int*   apos      = (const int*)d_in[1];
  const int*   btypes    = (const int*)d_in[2];
  const float* coords    = (const float*)d_in[3];
  const int*   eidx      = (const int*)d_in[4];
  const int*   bedges    = (const int*)d_in[5];
  const float* emb_atom  = (const float*)d_in[6];
  const float* emb_pos   = (const float*)d_in[7];
  const float* emb_block = (const float*)d_in[8];
  const float* rbf_freq  = (const float*)d_in[9];
  const float* inter_w1  = (const float*)d_in[10];
  const float* inter_b1  = (const float*)d_in[11];
  const float* inter_w2  = (const float*)d_in[12];
  const float* inter_b2  = (const float*)d_in[13];
  const float* upd_w1    = (const float*)d_in[14];
  const float* upd_b1    = (const float*)d_in[15];
  const float* upd_w2    = (const float*)d_in[16];
  const float* upd_b2    = (const float*)d_in[17];
  const float* edge_w1   = (const float*)d_in[18];
  const float* edge_b1   = (const float*)d_in[19];
  const float* edge_w2   = (const float*)d_in[20];
  const float* edge_b2   = (const float*)d_in[21];
  const float* attn_in_w = (const float*)d_in[22];
  const float* attn_in_b = (const float*)d_in[23];
  const float* attn_out_w= (const float*)d_in[24];
  const float* attn_out_b= (const float*)d_in[25];
  const float* ln_g      = (const float*)d_in[26];
  const float* ln_b      = (const float*)d_in[27];
  const float* fin_w1    = (const float*)d_in[28];
  const float* fin_b1    = (const float*)d_in[29];
  const float* fin_w2    = (const float*)d_in[30];
  const float* fin_b2    = (const float*)d_in[31];
  const float* fin_w3    = (const float*)d_in[32];
  const float* fin_b3    = (const float*)d_in[33];

  char* ws = (char*)d_ws;
  float*    x       = (float*)   (ws + OFF_X);
  _Float16* xh      = (_Float16*)(ws + OFF_XH);
  float*    agg     = (float*)   (ws + OFF_AGG);
  _Float16* rbfh    = (_Float16*)(ws + OFF_RBFH);
  _Float16* w1t     = (_Float16*)(ws + OFF_W1T);
  _Float16* w2t     = (_Float16*)(ws + OFF_W2T);
  _Float16* uw1t    = (_Float16*)(ws + OFF_UW1T);
  _Float16* uw2t    = (_Float16*)(ws + OFF_UW2T);
  _Float16* ew1t    = (_Float16*)(ws + OFF_EW1T);
  _Float16* ew2t    = (_Float16*)(ws + OFF_EW2T);
  _Float16* aiwt    = (_Float16*)(ws + OFF_AIWT);
  _Float16* aowt    = (_Float16*)(ws + OFF_AOWT);
  float*    blocks  = (float*)   (ws + OFF_BLOCKS);
  _Float16* blocksh = (_Float16*)(ws + OFF_BLOCKSH);
  _Float16* ef2h    = (_Float16*)(ws + OFF_EF2H);
  _Float16* qkvh    = (_Float16*)(ws + OFF_QKVH);
  _Float16* oh      = (_Float16*)(ws + OFF_OH);
  float*    ea      = (float*)   (ws + OFF_EA);
  float*    sums    = (float*)   (ws + OFF_SUMS);
  float*    cnt     = (float*)   (ws + OFF_CNT);
  _Float16* qkv2h   = (_Float16*)(ws + OFF_QKV2H);
  _Float16* oh2     = (_Float16*)(ws + OFF_OH2);
  float*    att     = (float*)   (ws + OFF_ATT);

  auto cvt = [&](const float* src, _Float16* dst, int K, int N, int Kpad) {
    int n = N * Kpad;
    k_convert_t<<<(n + 255) / 256, 256, 0, stream>>>(src, dst, K, N, Kpad);
  };

  // weight conversion (transpose -> [N][Kpad] f16)
  for (int b = 0; b < 2; ++b) {
    cvt(inter_w1 + (size_t)b * 262 * 128, w1t  + (size_t)b * 128 * KEF, 262, 128, KEF);
    cvt(inter_w2 + (size_t)b * 128 * 128, w2t  + (size_t)b * 128 * 128, 128, 128, 128);
    cvt(upd_w1   + (size_t)b * 256 * 128, uw1t + (size_t)b * 128 * 256, 256, 128, 256);
    cvt(upd_w2   + (size_t)b * 128 * 128, uw2t + (size_t)b * 128 * 128, 128, 128, 128);
  }
  cvt(edge_w1,    ew1t, 256, 128, 256);
  cvt(edge_w2,    ew2t, 128, 128, 128);
  cvt(attn_in_w,  aiwt, 128, 384, 128);
  cvt(attn_out_w, aowt, 128, 128, 128);

  // embeddings + RBF
  k_embed<<<(NATOM * HD + 255) / 256, 256, 0, stream>>>(A, apos, btypes, emb_atom,
                                                        emb_pos, emb_block, x, xh);
  k_rbf<<<(NEDGE + 255) / 256, 256, 0, stream>>>(coords, eidx, eidx + NEDGE, rbf_freq, rbfh);

  // interaction blocks (WMMA edge MLP + scatter, WMMA update MLP + residual)
  for (int b = 0; b < 2; ++b) {
    k_zero<<<(NATOM * HD + 255) / 256, 256, 0, stream>>>(agg, NATOM * HD);
    k_edge_mlp<<<NEDGE / 64, 256, 0, stream>>>(
        xh, rbfh, eidx, eidx + NEDGE,
        w1t + (size_t)b * 128 * KEF, inter_b1 + b * HD,
        w2t + (size_t)b * 128 * 128, inter_b2 + b * HD, agg);
    k_update<<<NATOM / 64, 256, 0, stream>>>(
        agg, uw1t + (size_t)b * 128 * 256, upd_b1 + b * HD,
        uw2t + (size_t)b * 128 * 128, upd_b2 + b * HD, x, xh);
  }

  // block pooling
  k_pool<<<(NBLKS * HD + 255) / 256, 256, 0, stream>>>(x, blocks, blocksh);

  // block-edge attention
  k_blockedge_mlp<<<BSZ * EBE / 64, 256, 0, stream>>>(blocksh, bedges, ew1t, edge_b1,
                                                      ew2t, edge_b2, ef2h);
  k_gemm<<<BSZ * EBE / 64, 256, 0, stream>>>(ef2h, aiwt, attn_in_b, qkvh, nullptr, 384, 128);
  k_attn<<<BSZ * 8, 256, 0, stream>>>(qkvh, oh, EBE);
  k_gemm<<<BSZ * EBE / 64, 256, 0, stream>>>(oh, aowt, attn_out_b, nullptr, ea, 128, 128);

  k_zero<<<(NBLKS * HD + NBLKS + 255) / 256, 256, 0, stream>>>(sums, NBLKS * HD + NBLKS);
  k_cnt<<<(BSZ * EBE + 255) / 256, 256, 0, stream>>>(bedges, cnt);
  k_sums<<<(BSZ * EBE * HD + 255) / 256, 256, 0, stream>>>(bedges, ea, sums);
  k_blockadd<<<(NBLKS * HD + 255) / 256, 256, 0, stream>>>(sums, cnt, blocks, blocksh);

  // block self-attention + head
  k_gemm<<<NBLKS / 64, 256, 0, stream>>>(blocksh, aiwt, attn_in_b, qkv2h, nullptr, 384, 128);
  k_attn<<<BSZ * 8, 256, 0, stream>>>(qkv2h, oh2, LBLK);
  k_gemm<<<NBLKS / 64, 256, 0, stream>>>(oh2, aowt, attn_out_b, nullptr, att, 128, 128);
  k_final<<<BSZ, 128, 0, stream>>>(att, ln_g, ln_b, fin_w1, fin_b1, fin_w2, fin_b2,
                                   fin_w3, fin_b3, (float*)d_out);
}
